// IF4SR_14809047237198
// MI455X (gfx1250) — matrix operand order repaired and checked by reading, not compiled, not run on hardware
//
#include <hip/hip_runtime.h>
#include <math.h>

// ---------------------------------------------------------------------------
// MI455X (gfx1250, wave32) implementation.
// All dense GEMMs run on v_wmma_f32_16x16x32_bf16 (bf16 in, f32 accumulate).
// GEMM inner loop: wave-uniform interior fast path (no guards, no exec churn)
// + ping-pong double-buffered fragment loads (no register copies) so
// global_load_b128 overlaps the WMMAs.
// ---------------------------------------------------------------------------

typedef __attribute__((ext_vector_type(16))) __bf16        v16bf;
typedef __attribute__((ext_vector_type(8)))  float         v8f;
typedef __attribute__((ext_vector_type(8)))  unsigned int  v8u;

// problem dims
static constexpr int  B_   = 512;
static constexpr int  L_   = 200;
static constexpr int  H_   = 128;
static constexpr int  LPAD = 224;     // L padded to K%32==0
static constexpr int  SH_  = 512;
static constexpr int  FH_  = 256;
static constexpr int  D_   = 32;
static constexpr int  FT_  = 32;
static constexpr int  TP_  = 64;
static constexpr int  NT_  = 32768;   // B*TP
static constexpr int  NI_  = 102400;  // B*IP
static constexpr int  ITEMS_ = 100001;
static constexpr long BLH  = (long)B_ * L_ * H_;   // 13,107,200

// epilogue flags
#define FLAG_GELU  1
#define FLAG_RESID 2
#define FLAG_BF16  4
#define FLAG_TRANS 8
#define FLAG_NOF32 16

// ------------------------------- device helpers ----------------------------

__device__ __forceinline__ float gelu_f(float x) {
  return 0.5f * x * (1.0f + erff(x * 0.70710678118654752f));
}
__device__ __forceinline__ float lrelu_f(float x) { return x > 0.0f ? x : 0.2f * x; }

__device__ __forceinline__ v16bf zero_frag() {
  v8u r = {0u, 0u, 0u, 0u, 0u, 0u, 0u, 0u};
  return __builtin_bit_cast(v16bf, r);
}
// Load 16 bf16 as two 16B chunks at p and p+step (step in elements).
__device__ __forceinline__ v16bf load_pair(const __bf16* p, int step) {
  uint4 a = *(const uint4*)p;
  uint4 b = *(const uint4*)(p + step);
  v8u r = {a.x, a.y, a.z, a.w, b.x, b.y, b.z, b.w};
  return __builtin_bit_cast(v16bf, r);
}

// 4x4 tile of 16x16x32 WMMAs
__device__ __forceinline__ void mma16(v8f (&acc)[4][4], const v16bf (&fa)[4],
                                      const v16bf (&fb)[4]) {
#pragma unroll
  for (int i = 0; i < 4; i++)
#pragma unroll
    for (int j = 0; j < 4; j++)
      acc[i][j] = __builtin_amdgcn_wmma_f32_16x16x32_bf16(
          false, fa[i], false, fb[j], (short)0, acc[i][j], false, false);
}

__device__ __forceinline__ float block_sum_128(float v) {
  __shared__ float red[128];
  red[threadIdx.x] = v;
  __syncthreads();
  for (int s = 64; s > 0; s >>= 1) {
    if ((int)threadIdx.x < s) red[threadIdx.x] += red[threadIdx.x + s];
    __syncthreads();
  }
  float r = red[0];
  __syncthreads();
  return r;
}

// ------------------------------- WMMA GEMM ---------------------------------
// C[M,N](f32) = op( A[M,K](bf16,row-major, ld=lda) * Bt[N,K](bf16,row-major,
// ld=ldb) ) ; Bt is B transposed. Block = 128 threads = 4 waves, each wave a
// 64x64 tile (16 wmma accs). K must be a multiple of 32 (buffers are padded).

template <int FLAGS, bool GUARD>
__device__ __forceinline__ void gemm_epilogue(
    v8f (&acc)[4][4], float* __restrict__ C, __bf16* __restrict__ Cb,
    const float* __restrict__ R, int M, int N, int ldc, long co,
    int tileM, int tileN, int lm, int lh) {
#pragma unroll
  for (int i = 0; i < 4; i++) {
#pragma unroll
    for (int j = 0; j < 4; j++) {
      const int nn = tileN + j * 16 + lm;
#pragma unroll
      for (int r = 0; r < 8; r++) {
        const int mm = tileM + i * 16 + lh * 8 + r;  // C layout: vgpr r, lane half
        if (!GUARD || (mm < M && nn < N)) {
          float v = acc[i][j][r];
          if (FLAGS & FLAG_GELU) v = gelu_f(v);
          long idx = (FLAGS & FLAG_TRANS) ? ((long)nn * ldc + mm) : ((long)mm * ldc + nn);
          idx += co;
          if (FLAGS & FLAG_RESID) v += R[idx];
          if (!(FLAGS & FLAG_NOF32)) C[idx] = v;
          if (FLAGS & FLAG_BF16) Cb[idx] = (__bf16)v;
        }
      }
    }
  }
}

template <int FLAGS>
__global__ __launch_bounds__(128) void gemm_bf16_k(
    const __bf16* __restrict__ A, const __bf16* __restrict__ Bt,
    float* __restrict__ C, __bf16* __restrict__ Cb, const float* __restrict__ R,
    int M, int N, int K, int lda, int ldb, int ldc,
    long sA, long sBt, long sC) {
  const int lane = threadIdx.x & 31;
  const int wave = threadIdx.x >> 5;
  const int wm = wave & 1, wn = wave >> 1;
  const int tileM = blockIdx.y * 128 + wm * 64;
  const int tileN = blockIdx.x * 128 + wn * 64;
  const int lm = lane & 15, lh = lane >> 4;
  A  += (long)blockIdx.z * sA;
  Bt += (long)blockIdx.z * sBt;
  const long co = (long)blockIdx.z * sC;

  v8f acc[4][4];
#pragma unroll
  for (int i = 0; i < 4; i++)
#pragma unroll
    for (int j = 0; j < 4; j++)
#pragma unroll
      for (int e = 0; e < 8; e++) acc[i][j][e] = 0.0f;

  const bool interior = (tileM + 64 <= M) && (tileN + 64 <= N);  // wave-uniform

  if (interior) {
    // --- fast path: no guards; ping-pong buffers, no inter-iteration copies ---
    const __bf16* ap = A + (long)(tileM + lm) * lda + lh * 8;
    const __bf16* bp = Bt + (long)(tileN + lm) * ldb + lh * 8;
    v16bf fa0[4], fb0[4], fa1[4], fb1[4];
#pragma unroll
    for (int i = 0; i < 4; i++) fa0[i] = load_pair(ap + (long)i * 16 * lda, 16);
#pragma unroll
    for (int j = 0; j < 4; j++) fb0[j] = load_pair(bp + (long)j * 16 * ldb, 16);
    int k0 = 32;
    while (true) {
      // even half: prefetch k0 into buf1, compute on buf0
      bool more = (k0 < K);
      if (more) {
#pragma unroll
        for (int i = 0; i < 4; i++) fa1[i] = load_pair(ap + k0 + (long)i * 16 * lda, 16);
#pragma unroll
        for (int j = 0; j < 4; j++) fb1[j] = load_pair(bp + k0 + (long)j * 16 * ldb, 16);
      }
      mma16(acc, fa0, fb0);
      if (!more) break;
      k0 += 32;
      // odd half: prefetch k0 into buf0, compute on buf1
      more = (k0 < K);
      if (more) {
#pragma unroll
        for (int i = 0; i < 4; i++) fa0[i] = load_pair(ap + k0 + (long)i * 16 * lda, 16);
#pragma unroll
        for (int j = 0; j < 4; j++) fb0[j] = load_pair(bp + k0 + (long)j * 16 * ldb, 16);
      }
      mma16(acc, fa1, fb1);
      if (!more) break;
      k0 += 32;
    }
    gemm_epilogue<FLAGS, false>(acc, C, Cb, R, M, N, ldc, co, tileM, tileN, lm, lh);
  } else {
    // --- edge path: per-fragment guards ---
    for (int k0 = 0; k0 < K; k0 += 32) {
      v16bf fa[4], fb[4];
#pragma unroll
      for (int i = 0; i < 4; i++) {
        const int m = tileM + i * 16 + lm;
        // A fragment: lanes 0-15 -> K {0..7,16..23}, lanes 16-31 -> {8..15,24..31}
        fa[i] = (m < M) ? load_pair(A + (long)m * lda + k0 + lh * 8, 16) : zero_frag();
      }
#pragma unroll
      for (int j = 0; j < 4; j++) {
        const int n = tileN + j * 16 + lm;
        // B fragment from Bt: lanes 0-15 -> K {0..7,16..23}, lanes 16-31 -> {8..15,24..31}
        fb[j] = (n < N) ? load_pair(Bt + (long)n * ldb + k0 + lh * 8, 16) : zero_frag();
      }
      mma16(acc, fa, fb);
    }
    gemm_epilogue<FLAGS, true>(acc, C, Cb, R, M, N, ldc, co, tileM, tileN, lm, lh);
  }
}

// ------------------------------ small kernels ------------------------------

__global__ void fill_bf16_k(__bf16* p, long n) {
  long i = (long)blockIdx.x * blockDim.x + threadIdx.x;
  if (i < n) p[i] = (__bf16)0.0f;
}
__global__ void cast_bf16_k(const float* __restrict__ s, __bf16* __restrict__ d, long n) {
  long i = (long)blockIdx.x * blockDim.x + threadIdx.x;
  if (i < n) d[i] = (__bf16)s[i];
}
// dst[n*Kpad + k] = (k<K) ? src[k*N + n] : 0   (f32 [K,N] -> bf16 [N,Kpad])
__global__ void wtrans_k(const float* __restrict__ src, __bf16* __restrict__ dst,
                         int K, int N, int Kpad) {
  long i = (long)blockIdx.x * blockDim.x + threadIdx.x;
  if (i >= (long)N * Kpad) return;
  int n = (int)(i / Kpad), k = (int)(i % Kpad);
  dst[i] = (k < K) ? (__bf16)src[(long)k * N + n] : (__bf16)0.0f;
}
__global__ void gather_f32_k(const float* __restrict__ emb, const int* __restrict__ ids,
                             float* __restrict__ out, long n) {
  long i = (long)blockIdx.x * blockDim.x + threadIdx.x;
  if (i >= n) return;
  out[i] = emb[((long)ids[i >> 7] << 7) + (i & 127)];
}
__global__ void gather_bf16_k(const float* __restrict__ emb, const int* __restrict__ ids,
                              __bf16* __restrict__ out, long n) {
  long i = (long)blockIdx.x * blockDim.x + threadIdx.x;
  if (i >= n) return;
  out[i] = (__bf16)emb[((long)ids[i >> 7] << 7) + (i & 127)];
}
__global__ void gather_tax_k(const float* __restrict__ emb, const int* __restrict__ ids,
                             float* __restrict__ taxh, __bf16* __restrict__ taxbf,
                             float* __restrict__ acc, long n) {
  long i = (long)blockIdx.x * blockDim.x + threadIdx.x;
  if (i >= n) return;
  float v = emb[((long)ids[i >> 7] << 7) + (i & 127)];
  taxh[i] = v;
  taxbf[i] = (__bf16)v;
  acc[i] = 0.0f;
}

// LayerNorm over H=128 (eps 1e-8). One block per row, 128 threads.
// doT=1: also write bf16 transposed [B,H,LPAD]; doT=0: write bf16 same layout.
__global__ void ln_k(const float* __restrict__ X, const float* __restrict__ gamma,
                     const float* __restrict__ beta, float* __restrict__ outF,
                     __bf16* __restrict__ outT, __bf16* __restrict__ outB, int doT) {
  long row = blockIdx.x;
  int h = threadIdx.x;
  float x = X[(row << 7) + h];
  float mean = block_sum_128(x) * (1.0f / 128.0f);
  float d = x - mean;
  float var = block_sum_128(d * d) * (1.0f / 128.0f);
  float y = d * rsqrtf(var + 1e-8f) * gamma[h] + beta[h];
  outF[(row << 7) + h] = y;
  if (doT) {
    long b = row / L_, l = row % L_;
    outT[((b << 7) + h) * LPAD + l] = (__bf16)y;
  } else {
    outB[(row << 7) + h] = (__bf16)y;
  }
}

// attention pooling: g[b,:] = sum_l softmax_l(V.wv) * V[b,l,:]
__global__ void attn_pool_k(const float* __restrict__ V, const float* __restrict__ wv,
                            float* __restrict__ g) {
  int b = blockIdx.x;
  __shared__ float sc[L_];
  __shared__ float ssum;
  const float* Vb = V + (long)b * L_ * H_;
  for (int l = threadIdx.x; l < L_; l += blockDim.x) {
    float d = 0.0f;
    for (int h = 0; h < H_; h++) d += Vb[l * H_ + h] * wv[h];
    sc[l] = d;
  }
  __syncthreads();
  if (threadIdx.x == 0) {
    float mx = -1e30f;
    for (int l = 0; l < L_; l++) mx = fmaxf(mx, sc[l]);
    float s = 0.0f;
    for (int l = 0; l < L_; l++) { sc[l] = expf(sc[l] - mx); s += sc[l]; }
    ssum = s;
  }
  __syncthreads();
  if (threadIdx.x < H_) {
    int h = threadIdx.x;
    float a = 0.0f;
    for (int l = 0; l < L_; l++) a += sc[l] * Vb[l * H_ + h];
    g[b * H_ + h] = a / ssum;
  }
}

// out[n,h4] = sum_d feat[n, h4*32+d] * vec[h4*32+d]
__global__ void gat_scores_k(const float* __restrict__ feat, const float* __restrict__ vec,
                             float* __restrict__ out, long n4) {
  long i = (long)blockIdx.x * blockDim.x + threadIdx.x;
  if (i >= n4) return;
  int node = (int)(i >> 2), h = (int)(i & 3);
  const float* f = feat + ((long)node << 7) + (h << 5);
  const float* v = vec + (h << 5);
  float s = 0.0f;
#pragma unroll
  for (int d = 0; d < 32; d++) s += f[d] * v[d];
  out[i] = s;
}

// Fused GAT hop: per-dst segment softmax (4 i2t edges, 2 t2t edges, contiguous),
// aggregate, residual+bias+relu for both edge types, sum, accumulate.
__global__ void gat_agg_k(const float* __restrict__ fsI, const float* __restrict__ ft,
                          const float* __restrict__ elI, const float* __restrict__ erI,
                          const float* __restrict__ elT, const float* __restrict__ erT,
                          const int* __restrict__ i2t_src, const int* __restrict__ t2t_src,
                          const float* __restrict__ bi, const float* __restrict__ bt,
                          float* __restrict__ taxh, __bf16* __restrict__ taxbf,
                          float* __restrict__ acc) {
  long i = (long)blockIdx.x * blockDim.x + threadIdx.x;
  if (i >= (long)NT_ * H_) return;
  int t = (int)(i >> 7), col = (int)(i & 127), h = col >> 5;
  float old = taxh[i];
  // i2t (4 in-edges)
  int s0 = i2t_src[t * 4 + 0], s1 = i2t_src[t * 4 + 1];
  int s2 = i2t_src[t * 4 + 2], s3 = i2t_src[t * 4 + 3];
  float er = erI[t * 4 + h];
  float e0 = lrelu_f(elI[s0 * 4 + h] + er), e1 = lrelu_f(elI[s1 * 4 + h] + er);
  float e2 = lrelu_f(elI[s2 * 4 + h] + er), e3 = lrelu_f(elI[s3 * 4 + h] + er);
  float mx = fmaxf(fmaxf(e0, e1), fmaxf(e2, e3));
  float a0 = expf(e0 - mx), a1 = expf(e1 - mx), a2 = expf(e2 - mx), a3 = expf(e3 - mx);
  float inv = 1.0f / (a0 + a1 + a2 + a3);
  float ti = (a0 * fsI[((long)s0 << 7) + col] + a1 * fsI[((long)s1 << 7) + col] +
              a2 * fsI[((long)s2 << 7) + col] + a3 * fsI[((long)s3 << 7) + col]) * inv;
  ti = fmaxf(ti + old + bi[col], 0.0f);
  // t2t (2 in-edges)
  int u0 = t2t_src[t * 2 + 0], u1 = t2t_src[t * 2 + 1];
  float er2 = erT[t * 4 + h];
  float f0 = lrelu_f(elT[u0 * 4 + h] + er2), f1 = lrelu_f(elT[u1 * 4 + h] + er2);
  float m2 = fmaxf(f0, f1);
  float b0 = expf(f0 - m2), b1 = expf(f1 - m2);
  float tt = (b0 * ft[((long)u0 << 7) + col] + b1 * ft[((long)u1 << 7) + col]) / (b0 + b1);
  tt = fmaxf(tt + old + bt[col], 0.0f);
  float nv = ti + tt;
  taxh[i] = nv;
  taxbf[i] = (__bf16)nv;
  acc[i] += nv;
}

// mul[b,f] = dot(local[b,f,:], g[b,:]);  local = 0.5*acc[root+b*TP] (valid)
__global__ void local_mul_k(const float* __restrict__ acc, const float* __restrict__ g,
                            const int* __restrict__ root, float* __restrict__ mul) {
  int bf = blockIdx.x;
  int b = bf >> 5;
  int r = root[bf];
  float v = 0.0f;
  if (r >= 0) {
    const float* row = acc + ((long)(b * TP_ + r) << 7);
    v = 0.5f * row[threadIdx.x] * g[b * H_ + threadIdx.x];
  }
  float s = block_sum_128(v);
  if (threadIdx.x == 0) mul[bf] = (r >= 0) ? s : 0.0f;
}

// per-b: LN over FT (all elems), mask zeros to -inf, softmax -> W  (one wave32)
__global__ void fuse_weight_k(const float* __restrict__ mul, const float* __restrict__ lg,
                              const float* __restrict__ lb, float* __restrict__ W) {
  int b = blockIdx.x, f = threadIdx.x;
  float x = mul[b * FT_ + f];
  float s = x;
  for (int m = 16; m > 0; m >>= 1) s += __shfl_xor(s, m, 32);
  float mean = s * (1.0f / 32.0f);
  float d = x - mean;
  float v = d * d;
  for (int m = 16; m > 0; m >>= 1) v += __shfl_xor(v, m, 32);
  v *= (1.0f / 32.0f);
  float ln = d * rsqrtf(v + 1e-8f) * lg[f] + lb[f];
  float masked = (x != 0.0f) ? ln : -__builtin_inff();
  float mx = masked;
  for (int m = 16; m > 0; m >>= 1) mx = fmaxf(mx, __shfl_xor(mx, m, 32));
  float e = expf(masked - mx);
  float se = e;
  for (int m = 16; m > 0; m >>= 1) se += __shfl_xor(se, m, 32);
  W[b * FT_ + f] = e / se;
}

// intention = LN(g + sum_f W*local) ; also bf16 copy for final GEMM
__global__ void fuse_intent_k(const float* __restrict__ g, const float* __restrict__ W,
                              const int* __restrict__ root, const float* __restrict__ acc,
                              const float* __restrict__ ig, const float* __restrict__ ib,
                              float* __restrict__ inten, __bf16* __restrict__ intenBf) {
  int b = blockIdx.x, h = threadIdx.x;
  float s = g[b * H_ + h];
  for (int f = 0; f < FT_; f++) {
    int r = root[b * FT_ + f];
    if (r >= 0) s += W[b * FT_ + f] * 0.5f * acc[((long)(b * TP_ + r) << 7) + h];
  }
  float mean = block_sum_128(s) * (1.0f / 128.0f);
  float d = s - mean;
  float var = block_sum_128(d * d) * (1.0f / 128.0f);
  float y = d * rsqrtf(var + 1e-8f) * ig[h] + ib[h];
  inten[b * H_ + h] = y;
  intenBf[b * H_ + h] = (__bf16)y;
}

// ------------------------------- host side ---------------------------------

extern "C" void kernel_launch(void* const* d_in, const int* in_sizes, int n_in,
                              void* d_out, int out_size, void* d_ws, size_t ws_size,
                              hipStream_t stream) {
  // inputs in setup_inputs() dict order
  const int*   seq       = (const int*)d_in[0];
  const int*   root      = (const int*)d_in[1];
  const int*   i2t_src   = (const int*)d_in[3];
  const int*   t2t_src   = (const int*)d_in[5];
  const int*   item_ids  = (const int*)d_in[6];
  const int*   tax_ids   = (const int*)d_in[7];
  const float* item_embed= (const float*)d_in[8];
  const float* tax_embed = (const float*)d_in[9];
  const float* scb_ln_g  = (const float*)d_in[10];
  const float* scb_ln_b  = (const float*)d_in[11];
  const float* scb_w1    = (const float*)d_in[12];
  const float* scb_w2    = (const float*)d_in[13];
  const float* fcb_ln_g  = (const float*)d_in[14];
  const float* fcb_ln_b  = (const float*)d_in[15];
  const float* fcb_w1    = (const float*)d_in[16];
  const float* fcb_w2    = (const float*)d_in[17];
  const float* fcb_w3    = (const float*)d_in[18];
  const float* wv        = (const float*)d_in[19];
  const float* gi_w      = (const float*)d_in[20];
  const float* gi_al     = (const float*)d_in[21];
  const float* gi_ar     = (const float*)d_in[22];
  const float* gi_b      = (const float*)d_in[23];
  const float* gt_w      = (const float*)d_in[24];
  const float* gt_al     = (const float*)d_in[25];
  const float* gt_ar     = (const float*)d_in[26];
  const float* gt_b      = (const float*)d_in[27];
  const float* liw_g     = (const float*)d_in[28];
  const float* liw_b     = (const float*)d_in[29];
  const float* int_g     = (const float*)d_in[30];
  const float* int_b     = (const float*)d_in[31];
  const float* uni_w     = (const float*)d_in[32];
  float* out = (float*)d_out;
  (void)in_sizes; (void)n_in; (void)out_size; (void)ws_size;

  // ---- carve workspace (~360 MB) ----
  size_t off = 0;
  auto carve = [&](size_t bytes) -> char* {
    char* p = (char*)d_ws + off;
    off += (bytes + 255) & ~(size_t)255;
    return p;
  };
  float*  V    = (float*)carve(BLH * 4);                 // [B,L,H]
  float*  F1   = (float*)carve(BLH * 4);                 // nv / nvs / fsI
  float*  F2   = (float*)carve(BLH * 4);                 // vs / fdI+ft
  __bf16* A1   = (__bf16*)carve((size_t)B_ * H_ * LPAD * 2);   // nv^T padded
  __bf16* Tbuf = (__bf16*)carve((size_t)B_ * H_ * SH_ * 2);    // t (bf16) / item_bf
  __bf16* Abf  = (__bf16*)carve(BLH * 2);                // nvs bf16
  __bf16* Bbf  = (__bf16*)carve(BLH * 2);                // z bf16 / tax_bf
  __bf16* Ych  = (__bf16*)carve((size_t)51200 * FH_ * 2);// fcb chunk
  float*  ACC  = (float*)carve((size_t)NT_ * H_ * 4);
  float*  TAXH = (float*)carve((size_t)NT_ * H_ * 4);
  __bf16* Ebf  = (__bf16*)carve((size_t)ITEMS_ * H_ * 2);
  __bf16* w1T[2], *w2T[2], *w1fT[2], *w2fT[2], *w3T[2], *giT[2], *gtT[2];
  for (int i = 0; i < 2; i++) {
    w1T[i]  = (__bf16*)carve((size_t)SH_ * LPAD * 2);
    w2T[i]  = (__bf16*)carve((size_t)L_ * SH_ * 2);
    w1fT[i] = (__bf16*)carve((size_t)FH_ * D_ * 2);
    w2fT[i] = (__bf16*)carve((size_t)D_ * FH_ * 2);
    w3T[i]  = (__bf16*)carve((size_t)H_ * H_ * 2);
    giT[i]  = (__bf16*)carve((size_t)H_ * H_ * 2);
    gtT[i]  = (__bf16*)carve((size_t)H_ * H_ * 2);
  }
  __bf16* uniT = (__bf16*)carve((size_t)H_ * H_ * 2);
  float* GB   = (float*)carve((size_t)B_ * H_ * 4);      // g
  float* ELI  = (float*)carve((size_t)NI_ * 4 * 4);
  float* ERI  = (float*)carve((size_t)NT_ * 4 * 4);
  float* ELT  = (float*)carve((size_t)NT_ * 4 * 4);
  float* ERT  = (float*)carve((size_t)NT_ * 4 * 4);
  float* MUL  = (float*)carve((size_t)B_ * FT_ * 4);
  float* WW   = (float*)carve((size_t)B_ * FT_ * 4);
  float* INTF = (float*)carve((size_t)B_ * H_ * 4);
  __bf16* INTB= (__bf16*)carve((size_t)B_ * H_ * 2);
  __bf16* PROJ= (__bf16*)carve((size_t)B_ * H_ * 2);
  float* FTb  = F2 + (long)NT_ * H_;   // t2t features live in upper half of F2

  const int T256 = 256;
  auto g1 = [](long n) { return dim3((unsigned)((n + 255) / 256)); };

  // ---- prep: zero A1 (K-pad tail), bf16 weights/tables ----
  long a1n = (long)B_ * H_ * LPAD;
  fill_bf16_k<<<g1(a1n), T256, 0, stream>>>(A1, a1n);
  cast_bf16_k<<<g1((long)ITEMS_ * H_), T256, 0, stream>>>(item_embed, Ebf, (long)ITEMS_ * H_);
  for (int i = 0; i < 2; i++) {
    wtrans_k<<<g1((long)SH_ * LPAD), T256, 0, stream>>>(scb_w1 + (long)i * L_ * SH_, w1T[i], L_, SH_, LPAD);
    wtrans_k<<<g1((long)L_ * SH_), T256, 0, stream>>>(scb_w2 + (long)i * SH_ * L_, w2T[i], SH_, L_, SH_);
    wtrans_k<<<g1((long)FH_ * D_), T256, 0, stream>>>(fcb_w1 + (long)i * D_ * FH_, w1fT[i], D_, FH_, D_);
    wtrans_k<<<g1((long)D_ * FH_), T256, 0, stream>>>(fcb_w2 + (long)i * FH_ * D_, w2fT[i], FH_, D_, FH_);
    wtrans_k<<<g1((long)H_ * H_), T256, 0, stream>>>(fcb_w3 + (long)i * H_ * H_, w3T[i], H_, H_, H_);
    wtrans_k<<<g1((long)H_ * H_), T256, 0, stream>>>(gi_w + (long)i * H_ * H_, giT[i], H_, H_, H_);
    wtrans_k<<<g1((long)H_ * H_), T256, 0, stream>>>(gt_w + (long)i * H_ * H_, gtT[i], H_, H_, H_);
  }
  wtrans_k<<<g1((long)H_ * H_), T256, 0, stream>>>(uni_w, uniT, H_, H_, H_);

  // ---- V = item_embed[seq] ----
  gather_f32_k<<<g1(BLH), T256, 0, stream>>>(item_embed, seq, V, BLH);

  // ---- mixer blocks ----
  for (int i = 0; i < 2; i++) {
    // LN -> nv(F1) + nv^T bf16 (A1, K padded to 224)
    ln_k<<<B_ * L_, 128, 0, stream>>>(V, scb_ln_g + i * H_, scb_ln_b + i * H_, F1, A1, nullptr, 1);
    // t = gelu(nv^T @ w1) : [B] x (128x224)@(224x512) -> bf16 T
    gemm_bf16_k<FLAG_GELU | FLAG_BF16 | FLAG_NOF32><<<dim3(4, 1, B_), 128, 0, stream>>>(
        A1, w1T[i], nullptr, Tbuf, nullptr, H_, SH_, LPAD, LPAD, LPAD, SH_,
        (long)H_ * LPAD, 0, (long)H_ * SH_);
    // vs = nv + (t @ w2)^T : [B] x (128x512)@(512x200), transposed store
    gemm_bf16_k<FLAG_RESID | FLAG_TRANS><<<dim3(2, 1, B_), 128, 0, stream>>>(
        Tbuf, w2T[i], F2, nullptr, F1, H_, L_, SH_, SH_, SH_, H_,
        (long)H_ * SH_, 0, (long)L_ * H_);
    // LN -> nvs(F1) + bf16 (Abf)
    ln_k<<<B_ * L_, 128, 0, stream>>>(F2, fcb_ln_g + i * H_, fcb_ln_b + i * H_, F1, nullptr, Abf, 0);
    // fcb: per-head MLP, chunked over rows (B*L*NH = 409600 rows of 32)
    for (int c = 0; c < 8; c++) {
      const long ro = (long)c * 51200;
      gemm_bf16_k<FLAG_GELU | FLAG_BF16 | FLAG_NOF32><<<dim3(2, 400, 1), 128, 0, stream>>>(
          Abf + ro * D_, w1fT[i], nullptr, Ych, nullptr, 51200, FH_, D_, D_, D_, FH_, 0, 0, 0);
      gemm_bf16_k<FLAG_BF16 | FLAG_NOF32><<<dim3(1, 400, 1), 128, 0, stream>>>(
          Ych, w2fT[i], nullptr, Bbf + ro * D_, nullptr, 51200, D_, FH_, FH_, FH_, D_, 0, 0, 0);
    }
    // V = nvs + z @ w3
    gemm_bf16_k<FLAG_RESID><<<dim3(1, 800, 1), 128, 0, stream>>>(
        Bbf, w3T[i], V, nullptr, F1, (int)(B_ * L_), H_, H_, H_, H_, H_, 0, 0, 0);
  }

  // ---- global intention g ----
  attn_pool_k<<<B_, 256, 0, stream>>>(V, wv, GB);

  // ---- GAT: gather node features ----
  gather_bf16_k<<<g1((long)NI_ * H_), T256, 0, stream>>>(item_embed, item_ids, Tbuf, (long)NI_ * H_);
  gather_tax_k<<<g1((long)NT_ * H_), T256, 0, stream>>>(tax_embed, tax_ids, TAXH, Bbf, ACC, (long)NT_ * H_);

  for (int hp = 0; hp < 2; hp++) {
    // fsI = item_h @ gi_w ; fdI = tax_h @ gi_w ; ft = tax_h @ gt_w
    gemm_bf16_k<0><<<dim3(1, 800, 1), 128, 0, stream>>>(
        Tbuf, giT[hp], F1, nullptr, nullptr, NI_, H_, H_, H_, H_, H_, 0, 0, 0);
    gemm_bf16_k<0><<<dim3(1, 256, 1), 128, 0, stream>>>(
        Bbf, giT[hp], F2, nullptr, nullptr, NT_, H_, H_, H_, H_, H_, 0, 0, 0);
    gemm_bf16_k<0><<<dim3(1, 256, 1), 128, 0, stream>>>(
        Bbf, gtT[hp], FTb, nullptr, nullptr, NT_, H_, H_, H_, H_, H_, 0, 0, 0);
    // attention scores per head
    gat_scores_k<<<g1((long)NI_ * 4), T256, 0, stream>>>(F1, gi_al + hp * H_, ELI, (long)NI_ * 4);
    gat_scores_k<<<g1((long)NT_ * 4), T256, 0, stream>>>(F2, gi_ar + hp * H_, ERI, (long)NT_ * 4);
    gat_scores_k<<<g1((long)NT_ * 4), T256, 0, stream>>>(FTb, gt_al + hp * H_, ELT, (long)NT_ * 4);
    gat_scores_k<<<g1((long)NT_ * 4), T256, 0, stream>>>(FTb, gt_ar + hp * H_, ERT, (long)NT_ * 4);
    // fused segment-softmax aggregation + residual + relu + accumulate
    gat_agg_k<<<g1((long)NT_ * H_), T256, 0, stream>>>(
        F1, FTb, ELI, ERI, ELT, ERT, i2t_src, t2t_src,
        gi_b + hp * H_, gt_b + hp * H_, TAXH, Bbf, ACC);
  }

  // ---- fuse global + local ----
  local_mul_k<<<B_ * FT_, 128, 0, stream>>>(ACC, GB, root, MUL);
  fuse_weight_k<<<B_, 32, 0, stream>>>(MUL, liw_g, liw_b, WW);
  fuse_intent_k<<<B_, 128, 0, stream>>>(GB, WW, root, ACC, int_g, int_b, INTF, INTB);

  // ---- logits = (intention @ uni_w) @ item_embed^T ----
  gemm_bf16_k<FLAG_BF16 | FLAG_NOF32><<<dim3(1, 4, 1), 128, 0, stream>>>(
      INTB, uniT, nullptr, PROJ, nullptr, B_, H_, H_, H_, H_, H_, 0, 0, 0);
  gemm_bf16_k<0><<<dim3(782, 4, 1), 128, 0, stream>>>(
      PROJ, Ebf, out, nullptr, nullptr, B_, ITEMS_, H_, H_, H_, ITEMS_, 0, 0, 0);
}
